// MMAttn_53919019434355
// MI455X (gfx1250) — compile-verified, matrix operand order checked by reading, not run on hardware
//
#include <hip/hip_runtime.h>
#include <hip/hip_bf16.h>
#include <cstdint>

// ---------------------------------------------------------------------------
// MI455X (gfx1250) fused multi-modal attention:
//   qkv GEMM (+bias +QKNorm +RoPE) -> flash attention -> out GEMM (+bias)
// Matmuls via v_wmma_f32_16x16x32_f16 (wave32, f32 accumulate).
// f16 tile staging via Tensor Data Mover (tensor_load_to_lds + s_wait_tensorcnt)
// where possible; all WMMA operand tiles are laid out k-contiguous in LDS so
// fragment fetches lower to ds_load_b128 (no scalar u16 gather).
// ---------------------------------------------------------------------------

typedef __attribute__((ext_vector_type(16))) _Float16 v16h;
typedef __attribute__((ext_vector_type(8)))  float    v8f;
typedef __attribute__((ext_vector_type(4)))  unsigned int u32x4;
typedef __attribute__((ext_vector_type(4)))  int      i32x4;
typedef __attribute__((ext_vector_type(8)))  int      i32x8;

#define B_      2
#define NS_     1024          // per-stream sequence length
#define NTOT_   2048          // concatenated sequence length
#define DM_     2048          // d_model
#define NH_     16
#define HD_     128           // head dim
#define QKVN_   6144          // 3*d_model

#if defined(__has_builtin)
#  if __has_builtin(__builtin_amdgcn_tensor_load_to_lds)
#    define HAVE_TDM 1
#  endif
#endif
#ifndef HAVE_TDM
#  define HAVE_TDM 0
#endif

__device__ __forceinline__ int lane_id() { return threadIdx.x & 31; }

#if HAVE_TDM
// Build a 2D TDM descriptor (D#) per CDNA5 ISA ch.8 and issue tensor_load_to_lds.
//   element size fixed at 2 bytes (f16); tile is tile_d0 x tile_d1 elements,
//   source row stride = stride0 elements; destination LDS is packed row-major.
__device__ __forceinline__ void tdm_load_2d_f16(const void* gsrc, unsigned lds_off,
                                                unsigned tensor_d0, unsigned tensor_d1,
                                                unsigned tile_d0, unsigned tile_d1,
                                                unsigned stride0) {
  const unsigned long long ga = (unsigned long long)(uintptr_t)gsrc;
  u32x4 g0;
  g0[0] = 1u;                                            // count=1, user-mode D#
  g0[1] = lds_off;                                       // lds_addr (bytes)
  g0[2] = (unsigned)(ga & 0xffffffffu);                  // global_addr[31:0]
  g0[3] = (unsigned)((ga >> 32) & 0x1ffffffu) | (2u << 30);  // addr[56:32] | type=2
  i32x8 g1;
  g1[0] = (int)(1u << 16);                               // wg_mask=0, data_size=1 (2B)
  g1[1] = (int)((tensor_d0 & 0xffffu) << 16);            // tensor_dim0[15:0] @ bit48
  g1[2] = (int)(((tensor_d0 >> 16) & 0xffffu) |
                ((tensor_d1 & 0xffffu) << 16));          // tensor_dim0 hi | dim1 lo
  g1[3] = (int)(((tensor_d1 >> 16) & 0xffffu) |
                ((tile_d0 & 0xffffu) << 16));            // tensor_dim1 hi | tile_dim0
  g1[4] = (int)(tile_d1 & 0xffffu);                      // tile_dim1 (tile_dim2=0)
  g1[5] = (int)stride0;                                  // tensor_dim0_stride[31:0]
  g1[6] = 0;
  g1[7] = 0;
  const i32x4 z4 = {0, 0, 0, 0};
#if __clang_major__ >= 23
  const i32x8 z8 = {0, 0, 0, 0, 0, 0, 0, 0};
  __builtin_amdgcn_tensor_load_to_lds(g0, g1, z4, z4, z8, 0);
#else
  __builtin_amdgcn_tensor_load_to_lds(g0, g1, z4, z4, 0);
#endif
}
#endif  // HAVE_TDM

__device__ __forceinline__ v8f wmma_f16(v16h a, v16h b, v8f c) {
  return __builtin_amdgcn_wmma_f32_16x16x32_f16(
      /*neg_a=*/false, a, /*neg_b=*/false, b,
      /*c_mod=*/(short)0, c, /*reuse_a=*/false, /*reuse_b=*/false);
}

// A fragment: 16x32 f16, per ISA 7.12.2 layout.
// element e of lane L:  m = L&15,  k = (e&7) + ((e>>3)<<4) + ((L>>4)<<3)
// With sk==1 the 8-element runs are contiguous -> ds_load_b128 pairs.
__device__ __forceinline__ v16h load_frag_a(const _Float16* p, int sm, int sk) {
  const int lane = lane_id();
  const int m  = lane & 15;
  const int kb = (lane >> 4) << 3;
  v16h f;
#pragma unroll
  for (int e = 0; e < 16; ++e) {
    const int k = (e & 7) + ((e >> 3) << 4) + kb;
    f[e] = p[m * sm + k * sk];
  }
  return f;
}

// B fragment: 32x16 f16 (K x N), cols n across lanes (mirror of A layout).
__device__ __forceinline__ v16h load_frag_b(const _Float16* p, int sk, int sn) {
  const int lane = lane_id();
  const int n  = lane & 15;
  const int kb = (lane >> 4) << 3;
  v16h f;
#pragma unroll
  for (int e = 0; e < 16; ++e) {
    const int k = (e & 7) + ((e >> 3) << 4) + kb;
    f[e] = p[k * sk + n * sn];
  }
  return f;
}

// C/D layout: VGPR r -> row r + 8*(lane>=16), col = lane&15.

// ---------------------------------------------------------------------------
// Kernel 1: QKV projection + bias + QKNorm + RoPE, f16 outputs.
// grid: x = 6144/128 col tiles, y = (B*NS)/64 row tiles. block = 256 (8 waves)
// Weight tile staged transposed (Blt[n][k]) so B-fragments are k-contiguous.
// ---------------------------------------------------------------------------
__global__ void qkv_kernel(const float* __restrict__ X,
                           const float* __restrict__ W,
                           const float* __restrict__ bias,
                           const float* __restrict__ qg,
                           const float* __restrict__ kg,
                           _Float16* __restrict__ Qb,
                           _Float16* __restrict__ Kb,
                           _Float16* __restrict__ Vb,
                           int seq_off) {
  const int nbase = blockIdx.x * 128;
  const int mbase = blockIdx.y * 64;
  const int tid = threadIdx.x;
  const int wv = tid >> 5;
  const int lane = tid & 31;

  __shared__ _Float16 Al[64 * 32];     // [m][k]
  __shared__ _Float16 Blt[128 * 32];   // [n][k]  (transposed weight tile)
  __shared__ float    St[64][128];

  v8f acc[4];
#pragma unroll
  for (int m = 0; m < 4; ++m)
#pragma unroll
    for (int e = 0; e < 8; ++e) acc[m][e] = 0.0f;

  for (int kt = 0; kt < DM_ / 32; ++kt) {
    const int k0 = kt * 32;
    // stage A (64x32) f32 -> f16
#pragma unroll
    for (int i = 0; i < 8; ++i) {
      const int idx = tid + i * 256;
      const int r = idx >> 5, k = idx & 31;
      Al[idx] = (_Float16)X[(size_t)(mbase + r) * DM_ + k0 + k];
    }
    // stage B (32x128) f32 -> f16, transposed into Blt[n][k]
    // (global side stays n-fast => coalesced; LDS side was scalar either way)
#pragma unroll
    for (int i = 0; i < 16; ++i) {
      const int idx = tid + i * 256;
      const int kr = idx >> 7, n = idx & 127;
      Blt[n * 32 + kr] = (_Float16)W[(size_t)(k0 + kr) * QKVN_ + nbase + n];
    }
    if (kt + 1 < DM_ / 32)
      __builtin_prefetch(&W[(size_t)(k0 + 32) * QKVN_ + nbase + (tid & 127)], 0, 1);
    __syncthreads();

    const v16h bf = load_frag_b(Blt + (wv * 16) * 32, /*sk=*/1, /*sn=*/32);
#pragma unroll
    for (int m = 0; m < 4; ++m) {
      const v16h af = load_frag_a(Al + m * 16 * 32, 32, 1);
      acc[m] = wmma_f16(af, bf, acc[m]);
    }
    __syncthreads();
  }

  // spill tile to LDS for the row-wise epilogue
#pragma unroll
  for (int m = 0; m < 4; ++m)
#pragma unroll
    for (int r = 0; r < 8; ++r)
      St[m * 16 + r + ((lane >> 4) << 3)][wv * 16 + (lane & 15)] = acc[m][r];
  __syncthreads();

  // bias add
#pragma unroll
  for (int i = 0; i < 32; ++i) {
    const int idx = tid + i * 256;
    St[idx >> 7][idx & 127] += bias[nbase + (idx & 127)];
  }
  __syncthreads();

  const int part = nbase >> 11;          // 0=Q, 1=K, 2=V
  const int head = (nbase & 2047) >> 7;  // head index within part
  if (tid < 64) {
    const int rg  = mbase + tid;
    const int bb  = rg >> 10;
    const int pos = (rg & 1023) + seq_off;          // concat-sequence position
    const size_t dst = (((size_t)(bb * NH_ + head)) * NTOT_ + pos) * HD_;
    if (part == 2) {
#pragma unroll 4
      for (int d = 0; d < HD_; ++d) Vb[dst + d] = (_Float16)St[tid][d];
    } else {
      const float* g = (part == 0) ? qg : kg;
      _Float16* dbuf = (part == 0) ? Qb : Kb;
      float ss = 0.0f;
#pragma unroll 4
      for (int d = 0; d < HD_; ++d) { const float v = St[tid][d]; ss += v * v; }
      const float inv = rsqrtf(ss * (1.0f / HD_) + 1e-6f);
#pragma unroll 2
      for (int i = 0; i < HD_ / 2; ++i) {
        const float x1 = St[tid][i]      * inv * g[i];
        const float x2 = St[tid][i + 64] * inv * g[i + 64];
        const float freq = __powf(10000.0f, -(float)(2 * i) * (1.0f / HD_));
        const float ang = (float)pos * freq;
        const float c = __cosf(ang), s = __sinf(ang);
        dbuf[dst + i]      = (_Float16)(x1 * c - x2 * s);
        dbuf[dst + i + 64] = (_Float16)(x1 * s + x2 * c);
      }
    }
  }
}

// ---------------------------------------------------------------------------
// Kernel 2: flash attention over the 2048-token concat sequence.
// grid: x = B*H (32), y = NTOT/128 (16). block = 256 (8 waves x 16 q-rows).
// K chunk staged by the Tensor Data Mover (row-major, fragments already
// k-contiguous); V chunk staged synchronously *transposed* (Vtt[d][key]) so
// the P*V B-fragments are also k-contiguous -> ds_load_b128.
// ---------------------------------------------------------------------------
__global__ void attn_kernel(const _Float16* __restrict__ Q,
                            const _Float16* __restrict__ K,
                            const _Float16* __restrict__ V,
                            _Float16* __restrict__ O) {
  const int bh = blockIdx.x;
  const int b = bh >> 4, h = bh & 15;
  const int qbase = blockIdx.y * 128;
  const int tid = threadIdx.x;
  const int wv = tid >> 5;
  const int lane = tid & 31;

  __shared__ _Float16 Kt[32 * 128];    // [key][d]
  __shared__ _Float16 Vtt[128 * 32];   // [d][key]  (transposed)
  __shared__ float    Ss[8][16 * 32];
  __shared__ _Float16 Ps[8][16 * 32];
  __shared__ float    m_s[128], l_s[128], sc_s[128];

#if HAVE_TDM
  const unsigned kt_off = (unsigned)(uintptr_t)(void*)Kt;  // LDS byte offset
#endif

  // per-wave Q fragments (16 rows x 128 head dim), kept in VGPRs all kernel
  const _Float16* Qp = Q + ((size_t)bh * NTOT_ + qbase + wv * 16) * HD_;
  v16h qf[4];
#pragma unroll
  for (int ks = 0; ks < 4; ++ks) qf[ks] = load_frag_a(Qp + ks * 32, HD_, 1);

  v8f o[8];
#pragma unroll
  for (int t = 0; t < 8; ++t)
#pragma unroll
    for (int e = 0; e < 8; ++e) o[t][e] = 0.0f;

  if (tid < 128) { m_s[tid] = -3.0e38f; l_s[tid] = 0.0f; }
  __syncthreads();

  const float sm_scale = 0.088388347648318447f;  // 1/sqrt(128)

  for (int kc = 0; kc < NTOT_ / 32; ++kc) {
    const int kb = kc * 32;
    const size_t src0 = ((size_t)bh * NTOT_ + kb) * HD_;
    // ---- stage K (TDM) and V (sync, transposed) ----
#if HAVE_TDM
    if (wv == 0)
      tdm_load_2d_f16(K + src0, kt_off, HD_, NTOT_, HD_, 32, HD_);
#else
#pragma unroll
    for (int i = 0; i < 16; ++i) {
      const int idx = tid + i * 256;
      Kt[idx] = K[src0 + (size_t)(idx >> 7) * HD_ + (idx & 127)];
    }
#endif
#pragma unroll
    for (int i = 0; i < 16; ++i) {
      const int idx = tid + i * 256;
      const int r = idx >> 7, d = idx & 127;       // global read stays d-fast
      Vtt[d * 32 + r] = V[src0 + (size_t)r * HD_ + d];
    }
#if HAVE_TDM
    if (wv == 0) __builtin_amdgcn_s_wait_tensorcnt(0);
#endif
    __syncthreads();

    // ---- S = Q * K^T (per-wave 16x32 tile, 2 col subtiles x 4 k-steps) ----
#pragma unroll
    for (int j = 0; j < 2; ++j) {
      v8f s;
#pragma unroll
      for (int e = 0; e < 8; ++e) s[e] = 0.0f;
#pragma unroll
      for (int ks = 0; ks < 4; ++ks) {
        const v16h bf = load_frag_b(Kt + (j * 16) * HD_ + ks * 32, 1, HD_);
        s = wmma_f16(qf[ks], bf, s);
      }
#pragma unroll
      for (int r = 0; r < 8; ++r)
        Ss[wv][(r + ((lane >> 4) << 3)) * 32 + j * 16 + (lane & 15)] =
            s[r] * sm_scale;
    }
    __syncthreads();

    // ---- online softmax: lanes 0..15 each own one query row ----
    if (lane < 16) {
      const int row = wv * 16 + lane;
      float mx = -3.0e38f;
#pragma unroll
      for (int c = 0; c < 32; ++c) mx = fmaxf(mx, Ss[wv][lane * 32 + c]);
      const float m_old = m_s[row];
      const float m_new = fmaxf(m_old, mx);
      float psum = 0.0f;
#pragma unroll
      for (int c = 0; c < 32; ++c) {
        const float p = __expf(Ss[wv][lane * 32 + c] - m_new);
        Ps[wv][lane * 32 + c] = (_Float16)p;
        psum += p;
      }
      const float corr = __expf(m_old - m_new);
      l_s[row]  = l_s[row] * corr + psum;
      sc_s[row] = corr;
      m_s[row]  = m_new;
    }
    __syncthreads();

    // ---- rescale O, then O += P * V ----
    {
      const int rowoff = wv * 16 + ((lane >> 4) << 3);
#pragma unroll
      for (int r = 0; r < 8; ++r) {
        const float sc = sc_s[rowoff + r];
#pragma unroll
        for (int t = 0; t < 8; ++t) o[t][r] *= sc;
      }
      const v16h af = load_frag_a(&Ps[wv][0], 32, 1);
#pragma unroll
      for (int t = 0; t < 8; ++t) {
        const v16h bf = load_frag_b(Vtt + (t * 16) * 32, /*sk=*/1, /*sn=*/32);
        o[t] = wmma_f16(af, bf, o[t]);
      }
    }
    __syncthreads();
  }

  // finalize: divide by l, write f16 to attn buffer [B, NTOT, DM]
  const int rowoff = wv * 16 + ((lane >> 4) << 3);
#pragma unroll
  for (int r = 0; r < 8; ++r) {
    const float linv = 1.0f / l_s[rowoff + r];
    const int token = qbase + rowoff + r;
#pragma unroll
    for (int t = 0; t < 8; ++t) {
      const size_t dst =
          ((size_t)b * NTOT_ + token) * DM_ + h * HD_ + t * 16 + (lane & 15);
      O[dst] = (_Float16)(o[t][r] * linv);
    }
  }
}

// ---------------------------------------------------------------------------
// Kernel 3: output projection + bias, f32 result.
// grid: x = DM/128 col tiles, y = (B*NS)/64 row tiles. block = 256.
// A-tile staging via TDM (overlapped with the f32->f16 transposed weight
// staging); all fragments k-contiguous.
// ---------------------------------------------------------------------------
__global__ void out_proj_kernel(const _Float16* __restrict__ A,
                                const float* __restrict__ W,
                                const float* __restrict__ bias,
                                float* __restrict__ out,
                                int seq_off) {
  const int nbase = blockIdx.x * 128;
  const int mbase = blockIdx.y * 64;
  const int tid = threadIdx.x;
  const int wv = tid >> 5;
  const int lane = tid & 31;

  __shared__ _Float16 Al[64 * 32];     // [m][k]
  __shared__ _Float16 Blt[128 * 32];   // [n][k]  (transposed weight tile)

#if HAVE_TDM
  const unsigned al_off = (unsigned)(uintptr_t)(void*)Al;
  // first row of this tile inside the f16 attention buffer [B, NTOT, DM]
  const size_t arow0 = (size_t)(mbase >> 10) * NTOT_ + (mbase & 1023) + seq_off;
#endif

  v8f acc[4];
#pragma unroll
  for (int m = 0; m < 4; ++m)
#pragma unroll
    for (int e = 0; e < 8; ++e) acc[m][e] = 0.0f;

  for (int kt = 0; kt < DM_ / 32; ++kt) {
    const int k0 = kt * 32;
    // ---- stage A (64x32 f16): TDM 2D tile, row stride DM_ ----
#if HAVE_TDM
    if (wv == 0)
      tdm_load_2d_f16(A + arow0 * DM_ + k0, al_off,
                      /*tensor_d0=*/DM_, /*tensor_d1=*/B_ * NTOT_,
                      /*tile_d0=*/32, /*tile_d1=*/64, /*stride0=*/DM_);
#else
#pragma unroll
    for (int i = 0; i < 8; ++i) {
      const int idx = tid + i * 256;
      const int r = idx >> 5, k = idx & 31;
      const int rg = mbase + r;
      const size_t arow = (size_t)(rg >> 10) * NTOT_ + (rg & 1023) + seq_off;
      Al[idx] = A[arow * DM_ + k0 + k];
    }
#endif
    // ---- stage B (32x128) f32 -> f16, transposed (overlaps the TDM) ----
#pragma unroll
    for (int i = 0; i < 16; ++i) {
      const int idx = tid + i * 256;
      const int kr = idx >> 7, n = idx & 127;
      Blt[n * 32 + kr] = (_Float16)W[(size_t)(k0 + kr) * DM_ + nbase + n];
    }
    if (kt + 1 < DM_ / 32)
      __builtin_prefetch(&W[(size_t)(k0 + 32) * DM_ + nbase + (tid & 127)], 0, 1);
#if HAVE_TDM
    if (wv == 0) __builtin_amdgcn_s_wait_tensorcnt(0);
#endif
    __syncthreads();

    const v16h bf = load_frag_b(Blt + (wv * 16) * 32, /*sk=*/1, /*sn=*/32);
#pragma unroll
    for (int m = 0; m < 4; ++m) {
      const v16h af = load_frag_a(Al + m * 16 * 32, 32, 1);
      acc[m] = wmma_f16(af, bf, acc[m]);
    }
    __syncthreads();
  }

  const int col = nbase + wv * 16 + (lane & 15);
  const float bv = bias[col];
#pragma unroll
  for (int m = 0; m < 4; ++m)
#pragma unroll
    for (int r = 0; r < 8; ++r) {
      const int row = mbase + m * 16 + r + ((lane >> 4) << 3);
      out[(size_t)row * DM_ + col] = acc[m][r] + bv;
    }
}

// ---------------------------------------------------------------------------
extern "C" void kernel_launch(void* const* d_in, const int* in_sizes, int n_in,
                              void* d_out, int out_size, void* d_ws, size_t ws_size,
                              hipStream_t stream) {
  const float* x1     = (const float*)d_in[0];
  const float* x2     = (const float*)d_in[1];
  const float* w_qkv1 = (const float*)d_in[2];
  const float* b_qkv1 = (const float*)d_in[3];
  const float* w_qkv2 = (const float*)d_in[4];
  const float* b_qkv2 = (const float*)d_in[5];
  const float* qg1    = (const float*)d_in[6];
  const float* kg1    = (const float*)d_in[7];
  const float* qg2    = (const float*)d_in[8];
  const float* kg2    = (const float*)d_in[9];
  const float* w_out1 = (const float*)d_in[10];
  const float* b_out1 = (const float*)d_in[11];
  const float* w_out2 = (const float*)d_in[12];
  const float* b_out2 = (const float*)d_in[13];

  const size_t qkv_elems = (size_t)B_ * NH_ * NTOT_ * HD_;  // 8.39M f16 each
  _Float16* Qb = (_Float16*)d_ws;
  _Float16* Kb = Qb + qkv_elems;
  _Float16* Vb = Kb + qkv_elems;
  _Float16* Ab = Vb + qkv_elems;  // [B, NTOT, DM] f16

  dim3 blk(256);
  qkv_kernel<<<dim3(QKVN_ / 128, (B_ * NS_) / 64), blk, 0, stream>>>(
      x1, w_qkv1, b_qkv1, qg1, kg1, Qb, Kb, Vb, /*seq_off=*/0);
  qkv_kernel<<<dim3(QKVN_ / 128, (B_ * NS_) / 64), blk, 0, stream>>>(
      x2, w_qkv2, b_qkv2, qg2, kg2, Qb, Kb, Vb, /*seq_off=*/NS_);

  attn_kernel<<<dim3(B_ * NH_, NTOT_ / 128), blk, 0, stream>>>(Qb, Kb, Vb, Ab);

  float* out = (float*)d_out;
  out_proj_kernel<<<dim3(DM_ / 128, (B_ * NS_) / 64), blk, 0, stream>>>(
      Ab, w_out1, b_out1, out, /*seq_off=*/0);
  out_proj_kernel<<<dim3(DM_ / 128, (B_ * NS_) / 64), blk, 0, stream>>>(
      Ab, w_out2, b_out2, out + (size_t)B_ * NS_ * DM_, /*seq_off=*/NS_);
}
